// SpatialAttention_83434034692534
// MI455X (gfx1250) — compile-verified, matrix operand order checked by reading, not run on hardware
//
#include <hip/hip_runtime.h>

// ---------------- types ----------------
typedef __attribute__((ext_vector_type(16))) __bf16 v16bf;
typedef __attribute__((ext_vector_type(8)))  __bf16 v8bf;
typedef __attribute__((ext_vector_type(8)))  float  v8f;

#define NB   32
#define WW   256
#define TT   1024
#define KTOT 8192   // 8*T

// Load a 16x32 bf16 A/B fragment row-slice from LDS (rows are 32 bf16 = 64B).
// lane: row = ln (within 16), k pattern: k0 = 8*(lane>>4): [k0..k0+7], [k0+16..k0+23]
__device__ __forceinline__ v16bf lds_frag(const __bf16* rowptr, int k0) {
    v8bf lo = *(const v8bf*)(rowptr + k0);
    v8bf hi = *(const v8bf*)(rowptr + k0 + 16);
    v16bf r;
    #pragma unroll
    for (int e = 0; e < 8; ++e) { r[e] = lo[e]; r[e + 8] = hi[e]; }
    return r;
}

// =====================================================================
// Kernel 1: energy[b, i, v] = sum_f Q[b,i,f] * K[b,f,v]
//   Q[b,i,f] = w1[(f>>8)&7] * x[b, f&255, 4i + (f>>11)] + b1[(f>>8)&7]
//   K[b,f,v] = w2[f>>10]    * x[b, (f>>2)&255, ((f&3)<<8)|v] + b2[f>>10]
// Workgroup: 256 threads (8 waves). WG tile: 64 (i) x 128 (v).
// Wave tile: 32x32 (2x2 WMMA tiles). K chunk = 32.
// =====================================================================
__global__ __launch_bounds__(256) void energy_kernel(
    const float* __restrict__ x,
    const float* __restrict__ w1, const float* __restrict__ b1,
    const float* __restrict__ w2, const float* __restrict__ b2,
    float* __restrict__ energy)
{
    __shared__ __bf16 Alds[64][32];    // [i][k]   (Q tile)
    __shared__ __bf16 Blds[128][32];   // [v][k]   (K tile, transposed)

    const int b      = blockIdx.x;
    const int i_base = blockIdx.y * 64;
    const int v_base = blockIdx.z * 128;

    const int tid  = threadIdx.x;
    const int lane = tid & 31;
    const int wid  = tid >> 5;
    const int wy   = wid >> 2;        // 0..1 : i sub-tile (32 rows)
    const int wx   = wid & 3;         // 0..3 : v sub-tile (32 cols)
    const int half = lane >> 4;
    const int ln   = lane & 15;
    const int k0   = half * 8;

    const float* __restrict__ xb = x + (size_t)b * WW * TT;

    v8f acc00 = {0.f,0.f,0.f,0.f,0.f,0.f,0.f,0.f};
    v8f acc01 = acc00, acc10 = acc00, acc11 = acc00;

    for (int kc = 0; kc < KTOT / 32; ++kc) {
        const int f0 = kc * 32;

        // ---- stage Q tile (A): Alds[0..63][0..31] ----
        {
            const int oc = (f0 >> 8) & 7;       // constant over chunk
            const int ac = f0 >> 11;            // constant over chunk
            const int w0 = f0 & 255;            // chunk-aligned
            const float s1 = w1[oc];
            const float o1 = b1[oc];
            const int ii = tid >> 2;            // 0..63
            const int u8 = (tid & 3) * 8;       // 0,8,16,24
            const int tq = 4 * (i_base + ii) + ac;
            const float* xp = xb + (size_t)(w0 + u8) * TT + tq;
            #pragma unroll
            for (int j = 0; j < 8; ++j) {
                float v = s1 * xp[(size_t)j * TT] + o1;
                Alds[ii][u8 + j] = (__bf16)v;
            }
        }

        // ---- stage K tile (B, transposed): Blds[0..127][0..31] ----
        {
            const int ok = f0 >> 10;            // constant over chunk
            const float s2 = w2[ok];
            const float o2 = b2[ok];
            const int u  = tid >> 3;            // 0..31 (f row)
            const int vs = (tid & 7) * 16;      // 0..112
            const int wk = ((f0 >> 2) + (u >> 2)) & 255;
            const int tk = ((u & 3) << 8) + v_base + vs;
            const float4* xp = (const float4*)(xb + (size_t)wk * TT + tk);
            #pragma unroll
            for (int q = 0; q < 4; ++q) {
                float4 c = xp[q];
                Blds[vs + q * 4 + 0][u] = (__bf16)(s2 * c.x + o2);
                Blds[vs + q * 4 + 1][u] = (__bf16)(s2 * c.y + o2);
                Blds[vs + q * 4 + 2][u] = (__bf16)(s2 * c.z + o2);
                Blds[vs + q * 4 + 3][u] = (__bf16)(s2 * c.w + o2);
            }
        }
        __syncthreads();

        // ---- fragments + 2x2 WMMA ----
        v16bf a0 = lds_frag(&Alds[wy * 32 + ln][0],      k0);
        v16bf a1 = lds_frag(&Alds[wy * 32 + 16 + ln][0], k0);
        v16bf b0 = lds_frag(&Blds[wx * 32 + ln][0],      k0);
        v16bf b1v = lds_frag(&Blds[wx * 32 + 16 + ln][0], k0);
        acc00 = __builtin_amdgcn_wmma_f32_16x16x32_bf16(false, a0, false, b0,  (short)0, acc00, false, false);
        acc01 = __builtin_amdgcn_wmma_f32_16x16x32_bf16(false, a0, false, b1v, (short)0, acc01, false, false);
        acc10 = __builtin_amdgcn_wmma_f32_16x16x32_bf16(false, a1, false, b0,  (short)0, acc10, false, false);
        acc11 = __builtin_amdgcn_wmma_f32_16x16x32_bf16(false, a1, false, b1v, (short)0, acc11, false, false);
        __syncthreads();
    }

    // ---- store 4x 16x16 f32 tiles ----
    const int i0 = i_base + wy * 32 + half * 8;
    const int v0 = v_base + wx * 32 + ln;
    float* eb = energy + (size_t)b * WW * WW;
    #pragma unroll
    for (int r = 0; r < 8; ++r) {
        eb[(size_t)(i0 + r)      * WW + v0]      = acc00[r];
        eb[(size_t)(i0 + r)      * WW + v0 + 16] = acc01[r];
        eb[(size_t)(i0 + 16 + r) * WW + v0]      = acc10[r];
        eb[(size_t)(i0 + 16 + r) * WW + v0 + 16] = acc11[r];
    }
}

// =====================================================================
// Kernel 2: per-row min/max normalize + softmax -> bf16 attention
// One 256-thread block per (b,i) row.
// =====================================================================
__global__ __launch_bounds__(256) void softmax_kernel(
    const float* __restrict__ energy, __bf16* __restrict__ att)
{
    const int row = blockIdx.x;          // b*W + i
    const int tid = threadIdx.x;
    const int wid = tid >> 5;

    __shared__ float smx[8], smn[8], ssm[8];

    float v = energy[(size_t)row * WW + tid];
    float mx = v, mn = v;
    #pragma unroll
    for (int off = 16; off > 0; off >>= 1) {
        mx = fmaxf(mx, __shfl_xor(mx, off, 32));
        mn = fminf(mn, __shfl_xor(mn, off, 32));
    }
    if ((tid & 31) == 0) { smx[wid] = mx; smn[wid] = mn; }
    __syncthreads();
    mx = smx[0]; mn = smn[0];
    #pragma unroll
    for (int i = 1; i < 8; ++i) { mx = fmaxf(mx, smx[i]); mn = fminf(mn, smn[i]); }

    const float norm = (v - mn) / (mx - mn + 1e-8f);
    const float p = __expf(norm - 1.0f);   // shift by ~row max (==1 after norm)

    float s = p;
    #pragma unroll
    for (int off = 16; off > 0; off >>= 1) s += __shfl_xor(s, off, 32);
    if ((tid & 31) == 0) ssm[wid] = s;
    __syncthreads();
    s = ssm[0];
    #pragma unroll
    for (int i = 1; i < 8; ++i) s += ssm[i];

    att[(size_t)row * WW + tid] = (__bf16)(p / s);
}

// =====================================================================
// Kernel 3: out[b,i,t] = relu(w_out * sum_j att[b,i,j]*x[b,j,t] + b_out)
//                        + x[b,i,t]
// Workgroup: 256 threads (8 waves). WG tile: 64 (i) x 128 (t).
// Wave tile: 32x32. K = 256 (8 chunks of 32).
// =====================================================================
__global__ __launch_bounds__(256) void out_kernel(
    const __bf16* __restrict__ att,
    const float* __restrict__ x,
    const float* __restrict__ w_out, const float* __restrict__ b_out,
    float* __restrict__ out)
{
    __shared__ __bf16 Alds[64][32];    // [i][j]
    __shared__ __bf16 Blds[128][32];   // [t][j] (x tile, transposed)

    const int b      = blockIdx.x;
    const int i_base = blockIdx.y * 64;
    const int t_base = blockIdx.z * 128;

    const int tid  = threadIdx.x;
    const int lane = tid & 31;
    const int wid  = tid >> 5;
    const int wy   = wid >> 2;
    const int wx   = wid & 3;
    const int half = lane >> 4;
    const int ln   = lane & 15;
    const int k0   = half * 8;

    const float*  __restrict__ xb = x   + (size_t)b * WW * TT;
    const __bf16* __restrict__ ab = att + (size_t)b * WW * WW;

    v8f acc00 = {0.f,0.f,0.f,0.f,0.f,0.f,0.f,0.f};
    v8f acc01 = acc00, acc10 = acc00, acc11 = acc00;

    for (int kc = 0; kc < WW / 32; ++kc) {
        const int j0 = kc * 32;

        // stage attention tile (already bf16): 16-byte copies
        {
            const int ii = tid >> 2;            // 0..63
            const int u8 = (tid & 3) * 8;       // 0,8,16,24
            *(v8bf*)&Alds[ii][u8] =
                *(const v8bf*)&ab[(size_t)(i_base + ii) * WW + j0 + u8];
        }
        // stage x tile, transposed, f32 -> bf16
        {
            const int u  = tid >> 3;            // 0..31 (j row)
            const int vs = (tid & 7) * 16;      // 0..112
            const float4* xp = (const float4*)(xb + (size_t)(j0 + u) * TT + t_base + vs);
            #pragma unroll
            for (int q = 0; q < 4; ++q) {
                float4 c = xp[q];
                Blds[vs + q * 4 + 0][u] = (__bf16)c.x;
                Blds[vs + q * 4 + 1][u] = (__bf16)c.y;
                Blds[vs + q * 4 + 2][u] = (__bf16)c.z;
                Blds[vs + q * 4 + 3][u] = (__bf16)c.w;
            }
        }
        __syncthreads();

        v16bf a0 = lds_frag(&Alds[wy * 32 + ln][0],      k0);
        v16bf a1 = lds_frag(&Alds[wy * 32 + 16 + ln][0], k0);
        v16bf b0 = lds_frag(&Blds[wx * 32 + ln][0],      k0);
        v16bf b1v = lds_frag(&Blds[wx * 32 + 16 + ln][0], k0);
        acc00 = __builtin_amdgcn_wmma_f32_16x16x32_bf16(false, a0, false, b0,  (short)0, acc00, false, false);
        acc01 = __builtin_amdgcn_wmma_f32_16x16x32_bf16(false, a0, false, b1v, (short)0, acc01, false, false);
        acc10 = __builtin_amdgcn_wmma_f32_16x16x32_bf16(false, a1, false, b0,  (short)0, acc10, false, false);
        acc11 = __builtin_amdgcn_wmma_f32_16x16x32_bf16(false, a1, false, b1v, (short)0, acc11, false, false);
        __syncthreads();
    }

    // fused epilogue: o = relu(w_out*acc + b_out) + x
    const float wo = w_out[0];
    const float bo = b_out[0];
    const int i0 = i_base + wy * 32 + half * 8;
    const int t0 = t_base + wx * 32 + ln;
    float* ob = out + (size_t)b * WW * TT;
    #pragma unroll
    for (int r = 0; r < 8; ++r) {
        {
            const size_t idx = (size_t)(i0 + r) * TT + t0;
            float o = wo * acc00[r] + bo;
            ob[idx] = fmaxf(o, 0.f) + xb[idx];
            float o2 = wo * acc01[r] + bo;
            ob[idx + 16] = fmaxf(o2, 0.f) + xb[idx + 16];
        }
        {
            const size_t idx = (size_t)(i0 + 16 + r) * TT + t0;
            float o = wo * acc10[r] + bo;
            ob[idx] = fmaxf(o, 0.f) + xb[idx];
            float o2 = wo * acc11[r] + bo;
            ob[idx + 16] = fmaxf(o2, 0.f) + xb[idx + 16];
        }
    }
}

// =====================================================================
extern "C" void kernel_launch(void* const* d_in, const int* in_sizes, int n_in,
                              void* d_out, int out_size, void* d_ws, size_t ws_size,
                              hipStream_t stream)
{
    const float* x     = (const float*)d_in[0];
    const float* w1    = (const float*)d_in[1];
    const float* b1    = (const float*)d_in[2];
    const float* w2    = (const float*)d_in[3];
    const float* b2    = (const float*)d_in[4];
    const float* w_out = (const float*)d_in[5];
    const float* b_out = (const float*)d_in[6];
    float* out = (float*)d_out;

    // workspace: energy f32 [32,256,256] then attention bf16 [32,256,256]
    float*  energy = (float*)d_ws;
    __bf16* att    = (__bf16*)((char*)d_ws + (size_t)NB * WW * WW * sizeof(float));

    energy_kernel<<<dim3(NB, WW / 64, WW / 128), 256, 0, stream>>>(
        x, w1, b1, w2, b2, energy);
    softmax_kernel<<<NB * WW, 256, 0, stream>>>(energy, att);
    out_kernel<<<dim3(NB, WW / 64, TT / 128), 256, 0, stream>>>(
        att, x, w_out, b_out, out);
}
